// RoPEAttention_4715874091362
// MI455X (gfx1250) — matrix operand order repaired and checked by reading, not compiled, not run on hardware
//
#include <hip/hip_runtime.h>

#define NSEQ 4096
#define CDIM 768
#define NH   12
#define HD   64

typedef __attribute__((ext_vector_type(16))) __bf16 v16bf;
typedef __attribute__((ext_vector_type(8)))  float  v8f;

union Frag16 { v16bf v; unsigned u[8]; };

__device__ __forceinline__ unsigned bfbits(float f) {
  union { float f; unsigned u; } x; x.f = f;
  return (x.u + 0x7FFFu + ((x.u >> 16) & 1u)) >> 16;   // RNE bf16
}

#if __has_builtin(__builtin_amdgcn_cvt_pk_bf16_f32)
typedef __attribute__((ext_vector_type(2))) __bf16 v2bf;
__device__ __forceinline__ unsigned packbf2(float lo, float hi) {
  union { v2bf v; unsigned u; } c;
  c.v = __builtin_amdgcn_cvt_pk_bf16_f32(lo, hi);   // D[15:0]=lo, D[31:16]=hi
  return c.u;
}
#else
__device__ __forceinline__ unsigned packbf2(float lo, float hi) {
  return (bfbits(hi) << 16) | (bfbits(lo) & 0xFFFFu);
}
#endif

__device__ __forceinline__ v8f wmma_bf16(v16bf a, v16bf b, v8f c) {
  return __builtin_amdgcn_wmma_f32_16x16x32_bf16(false, a, false, b, (short)0, c,
                                                 false, false);
}

// A-operand (16xK row tile): lane half h -> K k+16h..k+16h+15 (contiguous)
__device__ __forceinline__ void load_a(Frag16& A, const unsigned* row, int k2, int lh) {
  #pragma unroll
  for (int i = 0; i < 4; ++i) {
    A.u[i]     = row[k2 + lh * 8 + i];
    A.u[4 + i] = row[k2 + lh * 8 + 4 + i];
  }
}
// B-operand: 8 consecutive dwords (lane holds one column, K contiguous)
__device__ __forceinline__ void load_b(Frag16& B, const unsigned* p) {
  #pragma unroll
  for (int i = 0; i < 8; ++i) B.u[i] = p[i];
}

// ---------------------------------------------------------------------------
// Kernel 0a: one-time f32 -> packed-bf16 conversion (2 elems -> 1 dword)
// ---------------------------------------------------------------------------
__global__ __launch_bounds__(256)
void cvt_bf16_kernel(const float* __restrict__ src, unsigned* __restrict__ dst,
                     int n2)
{
  int i = blockIdx.x * 256 + threadIdx.x;
  if (i < n2) dst[i] = packbf2(src[2 * i], src[2 * i + 1]);
}

// ---------------------------------------------------------------------------
// Kernel 0b: RoPE table: cs/sn[n*32 + j] = cos/sin(n * 10000^(-j/32))
// ---------------------------------------------------------------------------
__global__ __launch_bounds__(256)
void rope_tab_kernel(float* __restrict__ cs, float* __restrict__ sn)
{
  int i = blockIdx.x * 256 + threadIdx.x;   // < 4096*32
  int n = i >> 5, j = i & 31;
  float inv = __powf(10000.0f, -(float)j * (1.0f / 32.0f));
  float c, s;
  sincosf((float)n * inv, &c, &s);
  cs[i] = c;
  sn[i] = s;
}

// ---------------------------------------------------------------------------
// Kernel 1: qkv = x @ qkv_w^T (bf16 operands), fused RoPE, q scaled 1/sqrt(D).
// q,k -> [H][N][D] bf16; v -> transposed [H][D][N].
// One wave = 32-row x 64-col strip: 2 A-frags x 4 B-frags -> 8 WMMAs/k-step,
// so each B tile load feeds two WMMAs and weight re-read traffic is halved.
// ---------------------------------------------------------------------------
__global__ __launch_bounds__(128)
void qkv_rope_kernel(const unsigned* __restrict__ xb, const unsigned* __restrict__ wb,
                     const float* __restrict__ cs, const float* __restrict__ sn,
                     unsigned short* __restrict__ Qr, unsigned short* __restrict__ Kr,
                     unsigned short* __restrict__ Vt)
{
  const int lane = threadIdx.x & 31;
  const int ln = lane & 15, lh = lane >> 4;
  const int wave = blockIdx.x * 4 + (threadIdx.x >> 5);
  const int rt = wave & 127;   // 32-row tile (128)
  const int cs_i = wave >> 7;  // 64-col strip (36)

  v8f acc[2][4];
  #pragma unroll
  for (int m = 0; m < 2; ++m)
    #pragma unroll
    for (int t = 0; t < 4; ++t)
      #pragma unroll
      for (int r = 0; r < 8; ++r) acc[m][t][r] = 0.f;

  const unsigned* xr[2];
  #pragma unroll
  for (int m = 0; m < 2; ++m)
    xr[m] = xb + (size_t)(rt * 32 + m * 16 + ln) * (CDIM / 2);
  const unsigned* wr[4];
  #pragma unroll
  for (int t = 0; t < 4; ++t)
    wr[t] = wb + (size_t)(cs_i * 64 + t * 16 + ln) * (CDIM / 2) + lh * 8;

  for (int k2 = 0; k2 < CDIM / 2; k2 += 16) {
    Frag16 A[2], B[4];
    #pragma unroll
    for (int m = 0; m < 2; ++m) load_a(A[m], xr[m], k2, lh);
    #pragma unroll
    for (int t = 0; t < 4; ++t) load_b(B[t], wr[t] + k2);
    #pragma unroll
    for (int t = 0; t < 4; ++t)
      #pragma unroll
      for (int m = 0; m < 2; ++m)
        acc[m][t] = wmma_bf16(A[m].v, B[t].v, acc[m][t]);
  }

  const int o0 = cs_i * 64;
  const int part = o0 / CDIM;          // 0=q 1=k 2=v
  const int h    = (o0 % CDIM) / HD;

  if (part == 2) {
    // Vt[(h*64+d)*N + n]: n contiguous over r -> pack 8 bf16, one b128 store
    #pragma unroll
    for (int m = 0; m < 2; ++m)
      #pragma unroll
      for (int t = 0; t < 4; ++t) {
        int d = t * 16 + ln;
        int n0 = rt * 32 + m * 16 + 8 * lh;
        uint4 p;
        p.x = packbf2(acc[m][t][0], acc[m][t][1]);
        p.y = packbf2(acc[m][t][2], acc[m][t][3]);
        p.z = packbf2(acc[m][t][4], acc[m][t][5]);
        p.w = packbf2(acc[m][t][6], acc[m][t][7]);
        *(uint4*)(Vt + (size_t)(h * HD + d) * NSEQ + n0) = p;
      }
  } else {
    unsigned short* dst = (part == 0) ? Qr : Kr;
    const float sc = (part == 0) ? 0.125f : 1.0f;
    #pragma unroll
    for (int m = 0; m < 2; ++m)
      #pragma unroll
      for (int t = 0; t < 4; ++t)
        #pragma unroll
        for (int r = 0; r < 8; ++r) {
          int n = rt * 32 + m * 16 + r + 8 * lh;
          int d = t * 16 + ln;
          float c = cs[n * 32 + (d & 31)];
          float s = sn[n * 32 + (d & 31)];
          float a = acc[m][t][r];
          float b = acc[m][(t + 2) & 3][r];   // partner d±32: frag t^±2
          float res = (t < 2) ? (a * c - b * s) : (a * c + b * s);
          dst[(size_t)(h * NSEQ + n) * HD + d] = (unsigned short)bfbits(res * sc);
        }
  }
}

// ---------------------------------------------------------------------------
// Kernel 2: flash attention per head. One wave owns 16 queries.
// S^T = K·Q^T (per-lane == per-query softmax), O^T = Vt·P^T.
// K and V fragments for the 32-key block load up front so V loads overlap the
// S^T WMMAs and the softmax VALU work.
// ---------------------------------------------------------------------------
__global__ __launch_bounds__(128)
void flash_attn_kernel(const unsigned short* __restrict__ Qr,
                       const unsigned short* __restrict__ Kr,
                       const unsigned short* __restrict__ Vt,
                       unsigned short* __restrict__ AO)
{
  const int lane = threadIdx.x & 31;
  const int ln = lane & 15, lh = lane >> 4;
  const int wave = blockIdx.x * 4 + (threadIdx.x >> 5);
  const int qt = wave & 255;
  const int h  = wave >> 8;

  Frag16 Bq[2];   // Q^T operands: dims 0..31 / 32..63
  {
    const unsigned* qrow =
        (const unsigned*)(Qr + (size_t)(h * NSEQ + qt * 16 + ln) * HD);
    #pragma unroll
    for (int j = 0; j < 2; ++j) {
      #pragma unroll
      for (int i = 0; i < 8; ++i) Bq[j].u[i] = qrow[j * 16 + (i >> 2) * 8 + lh * 4 + (i & 3)];
    }
  }

  v8f Ot[4];
  #pragma unroll
  for (int f = 0; f < 4; ++f)
    #pragma unroll
    for (int r = 0; r < 8; ++r) Ot[f][r] = 0.f;
  float m_i = -3.4e38f;
  float l_i = 0.f;

  const unsigned* kbase = (const unsigned*)(Kr + (size_t)h * NSEQ * HD);
  const unsigned* vbase = (const unsigned*)(Vt + (size_t)h * HD * NSEQ);

  for (int kb = 0; kb < NSEQ; kb += 32) {
    // ---- issue ALL loads for this 32-key block first ----
    Frag16 Ak[2][2];   // [key subtile][dim half]
    #pragma unroll
    for (int kt = 0; kt < 2; ++kt) {
      const unsigned* krow = kbase + (size_t)(kb + kt * 16 + ln) * (HD / 2);
      #pragma unroll
      for (int i = 0; i < 4; ++i) {
        Ak[kt][0].u[i]     = krow[lh * 4 + i];
        Ak[kt][0].u[4 + i] = krow[8 + lh * 4 + i];
        Ak[kt][1].u[i]     = krow[16 + lh * 4 + i];
        Ak[kt][1].u[4 + i] = krow[24 + lh * 4 + i];
      }
    }
    Frag16 Av[4];      // V^T tiles: M=dim, K=32 keys
    #pragma unroll
    for (int f = 0; f < 4; ++f) {
      const unsigned* vrow = vbase + (size_t)(16 * f + ln) * (NSEQ / 2) + kb / 2;
      #pragma unroll
      for (int i = 0; i < 4; ++i) {
        Av[f].u[i]     = vrow[lh * 8 + i];
        Av[f].u[4 + i] = vrow[lh * 8 + 4 + i];
      }
    }
    if (kb + 32 < NSEQ) {
      __builtin_prefetch(kbase + (size_t)(kb + 32 + ln) * (HD / 2), 0, 0);
      __builtin_prefetch(vbase + (size_t)ln * (NSEQ / 2) + (kb + 32) / 2, 0, 0);
    }

    // ---- S^T = K · Q^T ----
    v8f St[2];
    #pragma unroll
    for (int kt = 0; kt < 2; ++kt) {
      v8f z;
      #pragma unroll
      for (int r = 0; r < 8; ++r) z[r] = 0.f;
      v8f t0 = wmma_bf16(Ak[kt][0].v, Bq[0].v, z);
      St[kt] = wmma_bf16(Ak[kt][1].v, Bq[1].v, t0);
    }

    // ---- online softmax (per-lane == per-query) ----
    float mloc = -3.4e38f;
    #pragma unroll
    for (int f = 0; f < 2; ++f)
      #pragma unroll
      for (int r = 0; r < 8; ++r) mloc = fmaxf(mloc, St[f][r]);
    mloc = fmaxf(mloc, __shfl_xor(mloc, 16, 32));
    float mnew  = fmaxf(m_i, mloc);
    float alpha = __expf(m_i - mnew);

    float ps0[8], ps1[8];
    float lloc = 0.f;
    #pragma unroll
    for (int r = 0; r < 8; ++r) { ps0[r] = __expf(St[0][r] - mnew); lloc += ps0[r]; }
    #pragma unroll
    for (int r = 0; r < 8; ++r) { ps1[r] = __expf(St[1][r] - mnew); lloc += ps1[r]; }
    lloc += __shfl_xor(lloc, 16, 32);
    l_i = l_i * alpha + lloc;
    m_i = mnew;

    #pragma unroll
    for (int f = 0; f < 4; ++f)
      #pragma unroll
      for (int r = 0; r < 8; ++r) Ot[f][r] *= alpha;

    Frag16 Bp;   // P^T: half0 keys 0..15, half1 keys 16..31
    #pragma unroll
    for (int i = 0; i < 4; ++i) {
      float o0a = __shfl_xor(ps0[2*i],     16, 32);
      float o0b = __shfl_xor(ps0[2*i + 1], 16, 32);
      float o1a = __shfl_xor(ps1[2*i],     16, 32);
      float o1b = __shfl_xor(ps1[2*i + 1], 16, 32);
      float loA = lh ? o1a : ps0[2*i];
      float loB = lh ? o1b : ps0[2*i + 1];
      float hiA = lh ? ps1[2*i]     : o0a;
      float hiB = lh ? ps1[2*i + 1] : o0b;
      Bp.u[i]     = packbf2(loA, loB);
      Bp.u[4 + i] = packbf2(hiA, hiB);
    }

    // ---- O^T += Vt_tile · P^T ----
    #pragma unroll
    for (int f = 0; f < 4; ++f) Ot[f] = wmma_bf16(Av[f].v, Bp.v, Ot[f]);
  }

  // epilogue: d contiguous over r per lane -> packed b128 stores
  const float inv_l = 1.0f / l_i;
  const int q = qt * 16 + ln;
  #pragma unroll
  for (int f = 0; f < 4; ++f) {
    int d0 = 16 * f + 8 * lh;
    uint4 p;
    p.x = packbf2(Ot[f][0] * inv_l, Ot[f][1] * inv_l);
    p.y = packbf2(Ot[f][2] * inv_l, Ot[f][3] * inv_l);
    p.z = packbf2(Ot[f][4] * inv_l, Ot[f][5] * inv_l);
    p.w = packbf2(Ot[f][6] * inv_l, Ot[f][7] * inv_l);
    *(uint4*)(AO + (size_t)q * CDIM + h * HD + d0) = p;
  }
}

// ---------------------------------------------------------------------------
// Kernel 3: out = AO @ proj_w^T + proj_b (f32 out). One wave = 32x64 strip
// (2 A-frags x 4 B-frags -> 8 WMMAs per k-step).
// ---------------------------------------------------------------------------
__global__ __launch_bounds__(128)
void proj_kernel(const unsigned short* __restrict__ AO,
                 const unsigned* __restrict__ wb, const float* __restrict__ bias,
                 float* __restrict__ out)
{
  const int lane = threadIdx.x & 31;
  const int ln = lane & 15, lh = lane >> 4;
  const int wave = blockIdx.x * 4 + (threadIdx.x >> 5);
  const int rt = wave & 127;   // 32-row tile (128)
  const int ct = wave >> 7;    // 64-col strip (12)

  v8f acc[2][4];
  #pragma unroll
  for (int m = 0; m < 2; ++m)
    #pragma unroll
    for (int t = 0; t < 4; ++t)
      #pragma unroll
      for (int r = 0; r < 8; ++r) acc[m][t][r] = 0.f;

  const unsigned* arow[2];
  #pragma unroll
  for (int m = 0; m < 2; ++m)
    arow[m] = (const unsigned*)(AO + (size_t)(rt * 32 + m * 16 + ln) * CDIM);
  const unsigned* wr[4];
  #pragma unroll
  for (int t = 0; t < 4; ++t)
    wr[t] = wb + (size_t)(ct * 64 + t * 16 + ln) * (CDIM / 2) + lh * 8;

  for (int k2 = 0; k2 < CDIM / 2; k2 += 16) {
    Frag16 A[2], B[4];
    #pragma unroll
    for (int m = 0; m < 2; ++m) load_a(A[m], arow[m], k2, lh);
    #pragma unroll
    for (int t = 0; t < 4; ++t) load_b(B[t], wr[t] + k2);
    #pragma unroll
    for (int t = 0; t < 4; ++t)
      #pragma unroll
      for (int m = 0; m < 2; ++m)
        acc[m][t] = wmma_bf16(A[m].v, B[t].v, acc[m][t]);
  }

  #pragma unroll
  for (int m = 0; m < 2; ++m)
    #pragma unroll
    for (int t = 0; t < 4; ++t) {
      int col = ct * 64 + t * 16 + ln;
      float b = bias[col];
      #pragma unroll
      for (int r = 0; r < 8; ++r)
        out[(size_t)(rt * 32 + m * 16 + r + 8 * lh) * CDIM + col] = acc[m][t][r] + b;
    }
}

// ---------------------------------------------------------------------------
extern "C" void kernel_launch(void* const* d_in, const int* in_sizes, int n_in,
                              void* d_out, int out_size, void* d_ws, size_t ws_size,
                              hipStream_t stream)
{
  const float* x  = (const float*)d_in[0];   // [4096,768]
  const float* qw = (const float*)d_in[1];   // [2304,768]
  const float* pw = (const float*)d_in[2];   // [768,768]
  const float* pb = (const float*)d_in[3];   // [768]
  float* out = (float*)d_out;                // [4096,768] f32

  char* ws = (char*)d_ws;
  size_t off = 0;
  unsigned short* Qr = (unsigned short*)(ws + off); off += (size_t)NH*NSEQ*HD*2;   //  6.0 MB
  unsigned short* Kr = (unsigned short*)(ws + off); off += (size_t)NH*NSEQ*HD*2;   //  6.0 MB
  unsigned short* Vt = (unsigned short*)(ws + off); off += (size_t)NH*NSEQ*HD*2;   //  6.0 MB
  unsigned short* AO = (unsigned short*)(ws + off); off += (size_t)NSEQ*CDIM*2;    //  6.0 MB
  unsigned* xb  = (unsigned*)(ws + off); off += (size_t)NSEQ*CDIM*2;               //  6.0 MB
  unsigned* qwb = (unsigned*)(ws + off); off += (size_t)3*CDIM*CDIM*2;             //  3.4 MB
  unsigned* pwb = (unsigned*)(ws + off); off += (size_t)CDIM*CDIM*2;               //  1.1 MB
  float* csT = (float*)(ws + off); off += (size_t)NSEQ*32*4;                       //  0.5 MB
  float* snT = (float*)(ws + off); off += (size_t)NSEQ*32*4;                       //  0.5 MB

  int n2x = NSEQ * CDIM / 2, n2q = 3 * CDIM * CDIM / 2, n2p = CDIM * CDIM / 2;
  cvt_bf16_kernel<<<dim3((n2x + 255) / 256), dim3(256), 0, stream>>>(x,  xb,  n2x);
  cvt_bf16_kernel<<<dim3((n2q + 255) / 256), dim3(256), 0, stream>>>(qw, qwb, n2q);
  cvt_bf16_kernel<<<dim3((n2p + 255) / 256), dim3(256), 0, stream>>>(pw, pwb, n2p);
  rope_tab_kernel<<<dim3(NSEQ * 32 / 256), dim3(256), 0, stream>>>(csT, snT);

  // 128 row-tiles x 36 col-strips = 4608 waves / 4 per block
  qkv_rope_kernel<<<dim3(1152), dim3(128), 0, stream>>>(xb, qwb, csT, snT, Qr, Kr, Vt);
  // 256 query-tiles x 12 heads = 3072 waves / 4 per block
  flash_attn_kernel<<<dim3(768), dim3(128), 0, stream>>>(Qr, Kr, Vt, AO);
  // 128 row-tiles x 12 col-strips = 1536 waves / 4 per block
  proj_kernel<<<dim3(384), dim3(128), 0, stream>>>(AO, pwb, pb, out);

  (void)in_sizes; (void)n_in; (void)out_size; (void)ws_size;
}
